// MultiHeadAttention_32736240730518
// MI455X (gfx1250) — compile-verified, matrix operand order checked by reading, not run on hardware
//
#include <hip/hip_runtime.h>

// ---------------------------------------------------------------------------
// Multi-head attention for MI455X (gfx1250, wave32, WMMA).
//   0) convert Wq/Wk/Wv/Wo f32 -> f16 once (ws)
//   1) Q = query @ Wq^T + bq   -> f16 [B,H,S,64]       (ws)
//   2) K = key   @ Wk^T + bk   -> f16 [B,H,S,64]       (ws)
//   3) V = value @ Wv^T + bv   -> f16 [B,H,64,S] (T)   (ws)
//   4) flash attention (online softmax, async double-buffered K chunks)
//        -> ctx f16 [B,S,1024] (ws)
//   5) out = ctx @ Wo^T + bo   -> f32 [B,S,1024]       (d_out)
// All matmuls on v_wmma_f32_16x16x32_f16 (f32 accumulate). K chunks are
// streamed into LDS with global_load_async_to_lds_b128 / s_wait_asynccnt
// (CDNA5 async path) so the fetch overlaps the previous chunk's WMMA work;
// LDS reads are kept in addrspace(3) so they lower to ds_load_b128.
// f16 intermediates (40 MiB) stay resident in the 192 MB L2.
// ---------------------------------------------------------------------------

constexpr int DM   = 1024;      // d_model
constexpr int NH   = 16;        // heads
constexpr int DEP  = 64;        // head depth
constexpr int BB   = 2;         // batch
constexpr int SEQ  = 2048;      // sequence
constexpr int MTOK = BB * SEQ;  // 4096 tokens

typedef __attribute__((ext_vector_type(16))) _Float16 v16h;
typedef __attribute__((ext_vector_type(8)))  _Float16 v8h;
typedef __attribute__((ext_vector_type(8)))  float    v8f;
typedef __attribute__((ext_vector_type(4)))  float    v4f;

// ---- fragment loaders (ISA 7.12.2 16-bit 16x32 layout) ---------------------
// lane = row (mod 16); kh = lane>>4; elems 0..7 -> k = kh*8+0..7,
// elems 8..15 -> k = 16 + kh*8 + 0..7.  Two 16B vector loads per fragment.
__device__ inline v16h frag_rows_f16(const _Float16* p, int ld) {
  const int lane = threadIdx.x & 31;
  const int r = lane & 15, kh = lane >> 4;
  const _Float16* row = p + (long)r * ld + kh * 8;
  v8h lo = *(const v8h*)(row);
  v8h hi = *(const v8h*)(row + 16);
  v16h f;
#pragma unroll
  for (int i = 0; i < 8; ++i) { f[i] = lo[i]; f[8 + i] = hi[i]; }
  return f;
}

__device__ inline v16h frag_rows_f32(const float* p, int ld) {
  const int lane = threadIdx.x & 31;
  const int r = lane & 15, kh = lane >> 4;
  const float* row = p + (long)r * ld + kh * 8;
  v4f a0 = *(const v4f*)(row);
  v4f a1 = *(const v4f*)(row + 4);
  v4f b0 = *(const v4f*)(row + 16);
  v4f b1 = *(const v4f*)(row + 20);
  v16h f;
#pragma unroll
  for (int i = 0; i < 4; ++i) {
    f[i]      = (_Float16)a0[i];
    f[4 + i]  = (_Float16)a1[i];
    f[8 + i]  = (_Float16)b0[i];
    f[12 + i] = (_Float16)b1[i];
  }
  return f;
}

// ---- one-shot weight conversion: 4 x [1024x1024] f32 -> f16 ----------------
__global__ __launch_bounds__(256) void cvt_w4_kernel(
    const float* __restrict__ w0, const float* __restrict__ w1,
    const float* __restrict__ w2, const float* __restrict__ w3,
    _Float16* __restrict__ o0, _Float16* __restrict__ o1,
    _Float16* __restrict__ o2, _Float16* __restrict__ o3) {
  const long i = ((long)blockIdx.x * blockDim.x + threadIdx.x) * 8;
  const float* srcs[4] = {w0, w1, w2, w3};
  _Float16*    dsts[4] = {o0, o1, o2, o3};
#pragma unroll
  for (int m = 0; m < 4; ++m) {
    v4f x0 = *(const v4f*)(srcs[m] + i);
    v4f x1 = *(const v4f*)(srcs[m] + i + 4);
    v8h h;
#pragma unroll
    for (int j = 0; j < 4; ++j) {
      h[j]     = (_Float16)x0[j];
      h[4 + j] = (_Float16)x1[j];
    }
    *(v8h*)(dsts[m] + i) = h;
  }
}

// ---- GEMM: Y[m,n] = sum_k A[m,k] * Wh[n,k] + bias[n] -----------------------
// One 16x64 strip per wave (4 accumulator tiles, 4 wmma per K-step).
// OUT_MODE: 0 = f16 [B,H,S,64]; 1 = f16 [B,H,64,S]; 2 = f32 [MTOK, DM].
template <int A_F16, int OUT_MODE>
__global__ __launch_bounds__(256) void gemm_wmma_kernel(
    const void* __restrict__ Araw, const _Float16* __restrict__ Wh,
    const float* __restrict__ bias, void* __restrict__ out) {
  const int wave = (blockIdx.x * blockDim.x + threadIdx.x) >> 5;
  const int NSUP = DM / 64;                    // 16 super-tiles across N
  const int mt   = wave / NSUP;
  const int n0   = (wave % NSUP) * 64;
  const int lane = threadIdx.x & 31;

  v8f acc[4];
  v8f zero = {};
#pragma unroll
  for (int t = 0; t < 4; ++t) acc[t] = zero;

#pragma unroll 2
  for (int k0 = 0; k0 < DM; k0 += 32) {
    v16h a;
    if (A_F16) {
      a = frag_rows_f16((const _Float16*)Araw + (long)mt * 16 * DM + k0, DM);
    } else {
      const float* ap = (const float*)Araw + (long)mt * 16 * DM + k0;
      if (k0 + 32 < DM) __builtin_prefetch(ap + 32, 0, 1);  // global_prefetch
      a = frag_rows_f32(ap, DM);
    }
#pragma unroll
    for (int t = 0; t < 4; ++t) {
      v16h b = frag_rows_f16(Wh + (long)(n0 + t * 16) * DM + k0, DM);
      acc[t] = __builtin_amdgcn_wmma_f32_16x16x32_f16(false, a, false, b,
                                                      (short)0, acc[t],
                                                      false, false);
    }
  }

  const int nn    = lane & 15;
  const int mbase = mt * 16 + (lane >> 4) * 8;
#pragma unroll
  for (int t = 0; t < 4; ++t) {
    const int n = n0 + t * 16 + nn;
    const float bn = bias[n];
#pragma unroll
    for (int v = 0; v < 8; ++v) {
      const int m = mbase + v;
      const float y = acc[t][v] + bn;
      if (OUT_MODE == 2) {
        ((float*)out)[(long)m * DM + n] = y;
      } else {
        const int b_ = m / SEQ, s = m % SEQ;
        const int h = n >> 6, d = n & 63;
        long idx = (OUT_MODE == 0)
            ? ((long)(b_ * NH + h) * SEQ + s) * DEP + d    // [B,H,S,64]
            : ((long)(b_ * NH + h) * DEP + d) * SEQ + s;   // [B,H,64,S]
        ((_Float16*)out)[idx] = (_Float16)y;
      }
    }
  }
}

// ---- async helpers ---------------------------------------------------------
// One wave moves a 4 KB contiguous K chunk (32 rows x 64 f16) into LDS with 8
// async b128 transfers; the immediate offset applies to both global and LDS
// addresses (ISA 10.7 / 15.18.3).
#define ASYNC_K_SLICE(OFF)                                                    \
  asm volatile("global_load_async_to_lds_b128 %0, %1, off offset:" #OFF       \
               :: "v"(dst), "v"(src) : "memory")

__device__ inline void async_load_k_chunk(unsigned lds_base,
                                          const _Float16* gsrc) {
  const int lane = threadIdx.x & 31;
  unsigned dst = lds_base + lane * 16;            // LDS byte address
  const _Float16* src = gsrc + lane * 8;          // 16 B per lane
  ASYNC_K_SLICE(0);
  ASYNC_K_SLICE(512);
  ASYNC_K_SLICE(1024);
  ASYNC_K_SLICE(1536);
  ASYNC_K_SLICE(2048);
  ASYNC_K_SLICE(2560);
  ASYNC_K_SLICE(3072);
  ASYNC_K_SLICE(3584);
}

// ---- flash attention: one wave per 16-row Q tile, 4 waves / block ----------
__global__ __launch_bounds__(128) void attn_kernel(
    const _Float16* __restrict__ Qh, const _Float16* __restrict__ Kh,
    const _Float16* __restrict__ Vt, const float* __restrict__ mask,
    _Float16* __restrict__ Ctx) {
  __shared__ __align__(16) _Float16 ldsK[4][2][32 * DEP];  // 2x4KB per wave
  __shared__ __align__(16) _Float16 ldsP[4][16 * 32];      // 1KB per wave

  const int wslot = threadIdx.x >> 5;
  const int lane  = threadIdx.x & 31;
  const int wid   = blockIdx.x * (blockDim.x >> 5) + wslot;
  const int qtPerBH = SEQ / 16;                 // 128
  const int bh = wid / qtPerBH;
  const int qt = wid % qtPerBH;
  const int b  = bh / NH;
  const int h  = bh % NH;
  // LDS byte offsets for the async destinations (scalars only; the actual
  // LDS *reads* index the __shared__ arrays directly so they stay in
  // addrspace(3) and lower to ds_load_b128, not flat_load).
  const unsigned koff0 = (unsigned)(size_t)&ldsK[wslot][0][0];
  const unsigned koff1 = (unsigned)(size_t)&ldsK[wslot][1][0];

  // Q fragments for the whole tile (d = 0..31 and 32..63), loaded once.
  const _Float16* Qbase = Qh + ((long)bh * SEQ + qt * 16) * DEP;
  v16h qf0 = frag_rows_f16(Qbase, DEP);
  v16h qf1 = frag_rows_f16(Qbase + 32, DEP);

  const float* mbase_p = mask + (long)b * SEQ * SEQ;  // [B,1,S,S]
  const int g = lane >> 4, nn = lane & 15;
  const int q0 = qt * 16;

  float mrun[8], lrun[8];
  v8f acc[4];
  v8f zero = {};
#pragma unroll
  for (int v = 0; v < 8; ++v) { mrun[v] = -1e30f; lrun[v] = 0.f; }
#pragma unroll
  for (int t = 0; t < 4; ++t) acc[t] = zero;

  const _Float16* Kg = Kh + (long)bh * SEQ * DEP;   // chunks are contiguous
  constexpr int NCH = SEQ / 32;                     // 64 chunks

  // prime the pipeline: chunk 0 -> buffer 0
  async_load_k_chunk(koff0, Kg);

  for (int c = 0; c < NCH; ++c) {
    const int p = c & 1;
    if (c + 1 < NCH) {
      async_load_k_chunk(p ? koff0 : koff1, Kg + (long)(c + 1) * 32 * DEP);
      asm volatile("s_wait_asynccnt 0x8" ::: "memory");  // chunk c landed
    } else {
      asm volatile("s_wait_asynccnt 0x0" ::: "memory");
    }
    const int kv0 = c * 32;

    // ---- scores for a 16x32 chunk (two 16x16 tiles), K from LDS ----
    v8f s[2];
#pragma unroll
    for (int t = 0; t < 2; ++t) {
      v8f st = zero;
      const _Float16* Kt = &ldsK[wslot][p][t * 16 * DEP];  // stays addrspace(3)
      v16h kf0 = frag_rows_f16(Kt, DEP);         // B[k=d, n=kv] = K[kv, d]
      st = __builtin_amdgcn_wmma_f32_16x16x32_f16(false, qf0, false, kf0,
                                                  (short)0, st, false, false);
      v16h kf1 = frag_rows_f16(Kt + 32, DEP);
      st = __builtin_amdgcn_wmma_f32_16x16x32_f16(false, qf1, false, kf1,
                                                  (short)0, st, false, false);
#pragma unroll
      for (int v = 0; v < 8; ++v) {
        const int qrow = q0 + g * 8 + v;
        const int kcol = kv0 + t * 16 + nn;
        st[v] = st[v] * 0.125f + mbase_p[(long)qrow * SEQ + kcol] * -1e9f;
      }
      s[t] = st;
    }

    // ---- online softmax (row reductions across the 16-lane group) ----
    float scalef[8];
#pragma unroll
    for (int v = 0; v < 8; ++v) {
      float mx = fmaxf(s[0][v], s[1][v]);
#pragma unroll
      for (int off = 8; off >= 1; off >>= 1)
        mx = fmaxf(mx, __shfl_xor(mx, off, 32));
      const float mnew = fmaxf(mrun[v], mx);
      scalef[v] = __expf(mrun[v] - mnew);
      const float p0 = __expf(s[0][v] - mnew);
      const float p1 = __expf(s[1][v] - mnew);
      s[0][v] = p0; s[1][v] = p1;
      float rs = p0 + p1;
#pragma unroll
      for (int off = 8; off >= 1; off >>= 1)
        rs += __shfl_xor(rs, off, 32);
      lrun[v] = lrun[v] * scalef[v] + rs;
      mrun[v] = mnew;
    }

    // rescale running accumulators
#pragma unroll
    for (int t = 0; t < 4; ++t)
#pragma unroll
      for (int v = 0; v < 8; ++v) acc[t][v] *= scalef[v];

    // ---- stage P (C layout) -> LDS -> reload as A fragment ----
#pragma unroll
    for (int t = 0; t < 2; ++t)
#pragma unroll
      for (int v = 0; v < 8; ++v)
        ldsP[wslot][(g * 8 + v) * 32 + t * 16 + nn] = (_Float16)s[t][v];
    asm volatile("s_wait_dscnt 0" ::: "memory");
    v16h pa = frag_rows_f16(&ldsP[wslot][0], 32);

    // ---- acc += P @ V (V stored transposed: [B,H,64,S]) ----
#pragma unroll
    for (int t = 0; t < 4; ++t) {
      const _Float16* Vb = Vt + ((long)bh * DEP + t * 16) * (long)SEQ + kv0;
      v16h vf = frag_rows_f16(Vb, SEQ);          // B[k=kv, n=d] = Vt[d, kv]
      acc[t] = __builtin_amdgcn_wmma_f32_16x16x32_f16(false, pa, false, vf,
                                                      (short)0, acc[t],
                                                      false, false);
    }
  }

  // ---- normalize and store ctx as f16 [B,S,1024] ----
#pragma unroll
  for (int t = 0; t < 4; ++t) {
#pragma unroll
    for (int v = 0; v < 8; ++v) {
      const float y = acc[t][v] / lrun[v];
      const int m = q0 + g * 8 + v;
      const int d = h * DEP + t * 16 + nn;
      Ctx[((long)b * SEQ + m) * DM + d] = (_Float16)y;
    }
  }
}

// ---------------------------------------------------------------------------
extern "C" void kernel_launch(void* const* d_in, const int* in_sizes, int n_in,
                              void* d_out, int out_size, void* d_ws,
                              size_t ws_size, hipStream_t stream) {
  (void)in_sizes; (void)n_in; (void)out_size; (void)ws_size;
  const float* query = (const float*)d_in[0];
  const float* key   = (const float*)d_in[1];
  const float* value = (const float*)d_in[2];
  const float* mask  = (const float*)d_in[3];
  const float* Wq = (const float*)d_in[4];
  const float* bq = (const float*)d_in[5];
  const float* Wk = (const float*)d_in[6];
  const float* bk = (const float*)d_in[7];
  const float* Wv = (const float*)d_in[8];
  const float* bv = (const float*)d_in[9];
  const float* Wo = (const float*)d_in[10];
  const float* bo = (const float*)d_in[11];

  const size_t act = (size_t)MTOK * DM;          // 4096*1024 elements
  const size_t wel = (size_t)DM * DM;            // 1024*1024 elements
  _Float16* Qh  = (_Float16*)d_ws;               //  8 MiB
  _Float16* Kh  = Qh + act;                      //  8 MiB
  _Float16* Vt  = Kh + act;                      //  8 MiB
  _Float16* Ctx = Vt + act;                      //  8 MiB
  _Float16* Wqh = Ctx + act;                     //  2 MiB
  _Float16* Wkh = Wqh + wel;                     //  2 MiB
  _Float16* Wvh = Wkh + wel;                     //  2 MiB
  _Float16* Woh = Wvh + wel;                     //  2 MiB  (total 40 MiB)

  dim3 blk(256);

  // 0) weights f32 -> f16 (once)
  dim3 cvt_grid((unsigned)(wel / 8 / 256));      // 512 blocks
  cvt_w4_kernel<<<cvt_grid, blk, 0, stream>>>(Wq, Wk, Wv, Wo,
                                              Wqh, Wkh, Wvh, Woh);

  // 1-3) projections: each wave owns a 16x64 strip
  dim3 gemm_grid((MTOK / 16) * (DM / 64) / 8);   // 512 blocks (8 waves each)
  gemm_wmma_kernel<0, 0><<<gemm_grid, blk, 0, stream>>>(query, Wqh, bq, Qh);
  gemm_wmma_kernel<0, 0><<<gemm_grid, blk, 0, stream>>>(key,   Wkh, bk, Kh);
  gemm_wmma_kernel<0, 1><<<gemm_grid, blk, 0, stream>>>(value, Wvh, bv, Vt);

  // 4) attention: 4 waves per block (36 KB LDS/block)
  dim3 attn_blk(128);
  dim3 attn_grid(BB * NH * (SEQ / 16) / 4);      // 1024 blocks
  attn_kernel<<<attn_grid, attn_blk, 0, stream>>>(Qh, Kh, Vt, mask, Ctx);

  // 5) output projection (f16 A, f32 out)
  gemm_wmma_kernel<1, 2><<<gemm_grid, blk, 0, stream>>>(Ctx, Woh, bo, d_out);
}
